// GDN_64003602645026
// MI455X (gfx1250) — matrix operand order repaired and checked by reading, not compiled, not run on hardware
//
#include <hip/hip_runtime.h>
#include <math.h>

typedef __attribute__((ext_vector_type(16))) _Float16 v16h;
typedef __attribute__((ext_vector_type(8)))  _Float16 v8h;
typedef __attribute__((ext_vector_type(8)))  float    v8f;

#define NB_   16
#define NN_   2048
#define DD_   64
#define KK_   30
#define BN_   (NB_ * NN_)
#define EPS_  1e-5f
#define SLOPE_ 0.2f

__device__ __forceinline__ float wave_sum32(float v) {
#pragma unroll
  for (int off = 16; off > 0; off >>= 1) v += __shfl_xor(v, off, 32);
  return v;
}
__device__ __forceinline__ float wave_max32(float v) {
#pragma unroll
  for (int off = 16; off > 0; off >>= 1) v = fmaxf(v, __shfl_xor(v, off, 32));
  return v;
}

// ---------------- Kernel 1: rnorms + f16 staging of embeddings ----------
// One wave per row: computes 1/||w_row|| (single v_rsq_f32) and writes the
// row converted to f16 (wh, 256 KB, L2-resident) so the cos-sim GEMM reads
// f16 directly with b128 loads instead of f32 + cvt chains.
__global__ void __launch_bounds__(256) k_prep_emb(const float* __restrict__ w,
                                                  float* __restrict__ rn,
                                                  _Float16* __restrict__ wh) {
  const int wv = threadIdx.x >> 5, lane = threadIdx.x & 31;
  const int row = blockIdx.x * 8 + wv;
  const float v0 = w[row * DD_ + lane];
  const float v1 = w[row * DD_ + lane + 32];
  wh[row * DD_ + lane]      = (_Float16)v0;
  wh[row * DD_ + lane + 32] = (_Float16)v1;
  const float s = wave_sum32(v0 * v0 + v1 * v1);
  if (lane == 0) rn[row] = rsqrtf(s);
}

// ---------------- Kernel 1b: lin_w -> f16, transposed (column-major) ----
// lwt[n][k] = lin_w[k][n]; makes the B-fragment loads in k_xlin contiguous
// (b128) instead of 32 scattered dwords per tile. 8 KB, reused 2048x.
__global__ void __launch_bounds__(256) k_prep_lw(const float* __restrict__ lw,
                                                 _Float16* __restrict__ lwt) {
#pragma unroll
  for (int t = 0; t < 16; ++t) {
    const int idx = t * 256 + threadIdx.x;               // 4096 elements
    const int k = idx >> 6, n = idx & 63;
    lwt[n * DD_ + k] = (_Float16)lw[k * DD_ + n];
  }
}

// ---------------- Kernel 2: cos-sim strip (WMMA) + top-30 ---------------
// One workgroup (8 waves) per 16-row strip; the 16x2048 cos strip lives in
// 128 KB of the WGP's 320 KB LDS. A/B fragments load straight from the f16
// staging buffer (pure vector loads, zero converts in the hot loop).
__global__ void __launch_bounds__(256) k_costopk(const _Float16* __restrict__ wh,
                                                 const float* __restrict__ rn,
                                                 int* __restrict__ topk) {
  __shared__ float strip[16][NN_];                       // 131072 bytes
  const int wv   = threadIdx.x >> 5;
  const int lane = threadIdx.x & 31;
  const int lr   = lane & 15;
  const bool hi  = lane >= 16;
  const int row0 = blockIdx.x * 16;

  // A fragments: rows row0..row0+15, K = 0..63 (two 16x32 f16 fragments)
  const _Float16* arow = wh + (row0 + lr) * DD_;
  const int kbA = hi ? 16 : 0;
  const v8h aL0 = *reinterpret_cast<const v8h*>(arow + kbA);
  const v8h aH0 = *reinterpret_cast<const v8h*>(arow + kbA + 8);
  const v8h aL1 = *reinterpret_cast<const v8h*>(arow + 32 + kbA);
  const v8h aH1 = *reinterpret_cast<const v8h*>(arow + 32 + kbA + 8);
  v16h a0, a1;
#pragma unroll
  for (int j = 0; j < 8; ++j) {
    a0[j] = aL0[j]; a0[j + 8] = aH0[j];
    a1[j] = aL1[j]; a1[j + 8] = aH1[j];
  }
  // Row reciprocal norms hoisted out of the tile loop (reused 16x).
  float rinvr[8];
#pragma unroll
  for (int r = 0; r < 8; ++r) rinvr[r] = rn[row0 + r + (hi ? 8 : 0)];

  const int kbB = hi ? 16 : 0;
  for (int t = wv; t < NN_ / 16; t += 8) {
    const int c0 = t * 16;
    const _Float16* bcol = wh + (c0 + lr) * DD_;        // B(k,n) = w[n][k]
    const v16h b0 = *reinterpret_cast<const v16h*>(bcol + kbB);
    const v16h b1 = *reinterpret_cast<const v16h*>(bcol + 32 + kbB);
    v8f acc = {};
    acc = __builtin_amdgcn_wmma_f32_16x16x32_f16(false, a0, false, b0,
                                                 (short)0, acc, false, false);
    acc = __builtin_amdgcn_wmma_f32_16x16x32_f16(false, a1, false, b1,
                                                 (short)0, acc, false, false);
    const float rc = rn[c0 + lr];                       // 1/||w_col||
#pragma unroll
    for (int r = 0; r < 8; ++r) {
      const int m = r + (hi ? 8 : 0);
      strip[m][c0 + lr] = acc[r] * (rinvr[r] * rc);     // multiply-only
    }
  }
  __syncthreads();

  // Top-30 per row: wave wv owns rows wv and wv+8.
#pragma unroll
  for (int rr = 0; rr < 2; ++rr) {
    const int m = wv + rr * 8;
    float* rowv = strip[m];
    const int out_base = (row0 + m) * KK_;
    for (int sel = 0; sel < KK_; ++sel) {
      float best = -1e30f;
      int bi = 0x7fffffff;
      for (int j = lane; j < NN_; j += 32) {
        const float v = rowv[j];
        if (v > best) { best = v; bi = j; }
      }
#pragma unroll
      for (int off = 16; off > 0; off >>= 1) {
        const float ov = __shfl_xor(best, off, 32);
        const int   oi = __shfl_xor(bi, off, 32);
        if (ov > best || (ov == best && oi < bi)) { best = ov; bi = oi; }
      }
      if (lane == 0) {
        topk[out_base + sel] = bi;
        rowv[bi] = -1e30f;                              // mask winner
      }
    }
  }
}

// ---------------- Kernel 3: xl = x @ lin_w (WMMA) -----------------------
// A (x rows) converted inline (read exactly once); B from the f16
// transposed staging buffer -> contiguous vector loads.
__global__ void __launch_bounds__(256) k_xlin(const float* __restrict__ x,
                                              const _Float16* __restrict__ lwt,
                                              float* __restrict__ xl) {
  const int wv   = threadIdx.x >> 5;
  const int lane = threadIdx.x & 31;
  const int lr   = lane & 15;
  const bool hi  = lane >= 16;
  const int row0 = (blockIdx.x * 8 + wv) * 16;

  const float* arow = x + (row0 + lr) * DD_;
  const int kbA = hi ? 16 : 0;
  v16h a0, a1;
#pragma unroll
  for (int j = 0; j < 8; ++j) {
    a0[j]     = (_Float16)arow[kbA + j];
    a0[j + 8] = (_Float16)arow[kbA + 8 + j];
    a1[j]     = (_Float16)arow[32 + kbA + j];
    a1[j + 8] = (_Float16)arow[32 + kbA + 8 + j];
  }
  const int kbB = hi ? 16 : 0;

#pragma unroll
  for (int c0 = 0; c0 < DD_; c0 += 16) {
    const _Float16* bcol = lwt + (c0 + lr) * DD_;       // B(k,n) = lwt[n][k]
    const v16h b0 = *reinterpret_cast<const v16h*>(bcol + kbB);
    const v16h b1 = *reinterpret_cast<const v16h*>(bcol + 32 + kbB);
    v8f acc = {};
    acc = __builtin_amdgcn_wmma_f32_16x16x32_f16(false, a0, false, b0,
                                                 (short)0, acc, false, false);
    acc = __builtin_amdgcn_wmma_f32_16x16x32_f16(false, a1, false, b1,
                                                 (short)0, acc, false, false);
#pragma unroll
    for (int r = 0; r < 8; ++r)
      xl[(row0 + r + (hi ? 8 : 0)) * DD_ + c0 + lr] = acc[r];
  }
}

// ---------------- Kernel 4: per-node attention scores -------------------
__global__ void __launch_bounds__(256) k_scores(const float* __restrict__ xl,
                                                const float* __restrict__ emb,
                                                const float* __restrict__ att_i,
                                                const float* __restrict__ att_j,
                                                const float* __restrict__ att_em_i,
                                                const float* __restrict__ att_em_j,
                                                float* __restrict__ ai,
                                                float* __restrict__ aj) {
  const int wv = threadIdx.x >> 5, lane = threadIdx.x & 31;
  const int node = blockIdx.x * 8 + wv;
  const int i = node & (NN_ - 1);
  float si = 0.f, sj = 0.f;
#pragma unroll
  for (int t = 0; t < 2; ++t) {
    const int d = lane + 32 * t;
    const float xv = xl[node * DD_ + d];
    const float ev = emb[i * DD_ + d];
    si += xv * att_i[d] + ev * att_em_i[d];
    sj += xv * att_j[d] + ev * att_em_j[d];
  }
  si = wave_sum32(si);
  sj = wave_sum32(sj);
  if (lane == 0) { ai[node] = si; aj[node] = sj; }
}

// ---------------- Kernel 5: softmax-aggregate + fused readout -----------
__global__ void __launch_bounds__(256) k_aggregate(
    const float* __restrict__ xl, const int* __restrict__ topk,
    const float* __restrict__ ai, const float* __restrict__ aj,
    const float* __restrict__ emb, const float* __restrict__ gbias,
    const float* __restrict__ bn1g, const float* __restrict__ bn1b,
    const float* __restrict__ bn1m, const float* __restrict__ bn1v,
    const float* __restrict__ bn2g, const float* __restrict__ bn2b,
    const float* __restrict__ bn2m, const float* __restrict__ bn2v,
    const float* __restrict__ outw, const float* __restrict__ outb,
    float* __restrict__ y) {
  const int wv = threadIdx.x >> 5, lane = threadIdx.x & 31;
  const int node = blockIdx.x * 8 + wv;
  const int b = node >> 11;
  const int i = node & (NN_ - 1);

  int s = 0;
  float av = -1e30f;
  if (lane < KK_) {
    s = topk[i * KK_ + lane] + (b << 11);
    // Fire-and-forget prefetch (global_prefetch_b8): pull all 30 gather
    // rows toward the WGP while the softmax shuffle reductions run.
    __builtin_prefetch(xl + (size_t)s * DD_, 0, 0);
    const float v = ai[node] + aj[s];
    av = v > 0.f ? v : SLOPE_ * v;                       // leaky_relu(0.2)
  }
  const float mx = wave_max32(av);
  const float ex = (lane < KK_) ? __expf(av - mx) : 0.f;
  const float sum = wave_sum32(ex);
  const float alpha = ex / sum;

  float o0 = 0.f, o1 = 0.f;
#pragma unroll
  for (int k = 0; k < KK_; ++k) {
    const float ak = __shfl(alpha, k, 32);
    const int sk = __shfl(s, k, 32);
    const float* xr = xl + sk * DD_;                     // coalesced 256B row
    o0 = fmaf(ak, xr[lane], o0);
    o1 = fmaf(ak, xr[lane + 32], o1);
  }

  float p = 0.f;
#pragma unroll
  for (int t = 0; t < 2; ++t) {
    const int d = lane + 32 * t;
    float o = (t ? o1 : o0) + gbias[d];
    o = (o - bn1m[d]) * rsqrtf(bn1v[d] + EPS_) * bn1g[d] + bn1b[d];
    o = fmaxf(o, 0.f);
    float h = o * emb[i * DD_ + d];
    h = (h - bn2m[d]) * rsqrtf(bn2v[d] + EPS_) * bn2g[d] + bn2b[d];
    h = fmaxf(h, 0.f);
    p = fmaf(h, outw[d], p);
  }
  p = wave_sum32(p);
  if (lane == 0) y[node] = p + outb[0];
}

extern "C" void kernel_launch(void* const* d_in, const int* in_sizes, int n_in,
                              void* d_out, int out_size, void* d_ws, size_t ws_size,
                              hipStream_t stream) {
  const float* data   = (const float*)d_in[0];
  // d_in[1]: org_edge_index (unused, prior_graph path)
  const float* emb    = (const float*)d_in[2];
  const float* lw     = (const float*)d_in[3];
  const float* atti   = (const float*)d_in[4];
  const float* attj   = (const float*)d_in[5];
  const float* attemi = (const float*)d_in[6];
  const float* attemj = (const float*)d_in[7];
  const float* gbias  = (const float*)d_in[8];
  const float* bn1g   = (const float*)d_in[9];
  const float* bn1b   = (const float*)d_in[10];
  const float* bn1m   = (const float*)d_in[11];
  const float* bn1v   = (const float*)d_in[12];
  const float* bn2g   = (const float*)d_in[13];
  const float* bn2b   = (const float*)d_in[14];
  const float* bn2m   = (const float*)d_in[15];
  const float* bn2v   = (const float*)d_in[16];
  const float* outw   = (const float*)d_in[17];
  const float* outb   = (const float*)d_in[18];
  float* y = (float*)d_out;

  // Workspace carving (256B-aligned regions), total ~8.8 MB:
  char* ws = (char*)d_ws;
  float*    rn   = (float*)   (ws);                        //  2048 f32 (1/norm)
  int*      topk = (int*)     (ws + 8192);                 //  2048*30 i32
  _Float16* wh   = (_Float16*)(ws + 253952);               //  2048*64 f16
  _Float16* lwt  = (_Float16*)(ws + 516096);               //  64*64 f16 (transposed)
  float*    xl   = (float*)   (ws + 524288);               //  32768*64 f32
  float*    ai   = (float*)   (ws + 524288 + 8388608);     //  32768 f32
  float*    aj   = ai + BN_;                               //  32768 f32

  k_prep_emb<<<NN_ / 8,  256, 0, stream>>>(emb, rn, wh);
  k_prep_lw <<<1,        256, 0, stream>>>(lw, lwt);
  k_costopk <<<NN_ / 16, 256, 0, stream>>>(wh, rn, topk);
  k_xlin    <<<BN_ / 128, 256, 0, stream>>>(data, lwt, xl);
  k_scores  <<<BN_ / 8,  256, 0, stream>>>(xl, emb, atti, attj, attemi, attemj, ai, aj);
  k_aggregate<<<BN_ / 8, 256, 0, stream>>>(xl, topk, ai, aj, emb, gbias,
                                           bn1g, bn1b, bn1m, bn1v,
                                           bn2g, bn2b, bn2m, bn2v,
                                           outw, outb, y);
}
